// BaselineLSTM_5488968204527
// MI455X (gfx1250) — matrix operand order repaired and checked by reading, not compiled, not run on hardware
//
#include <hip/hip_runtime.h>
#include <math.h>

typedef __attribute__((ext_vector_type(16))) _Float16 v16h;
typedef __attribute__((ext_vector_type(8)))  float    v8f;

#define S_LEN   256
#define HID     64
#define NGATE   256      // 4*HID, PyTorch gate order i,f,g,o
#define NTILES  16       // NGATE/16
#define BT      32       // batch rows per workgroup
#define MT      2        // BT/16 (M tiles)
#define KH      2        // HID/32 k-steps for the recurrent matmul
#define NTPW    4        // n-tiles per wave (8 waves: 2 m-tiles x 4 n-groups)

__device__ __forceinline__ float sigm(float x) {
  return __frcp_rn(1.0f + __expf(-x));            // 1 transcendental + rcp
}
__device__ __forceinline__ float tanh_fast(float x) {
  return 1.0f - 2.0f * __frcp_rn(1.0f + __expf(2.0f * x));  // tanh via one exp
}

// Half-index of element (row m, k-within-frag kk) inside a 16x32 f16 A-fragment
// stored as [lane][16 halves] (this code's contiguous k order, matching the B frags).
__device__ __forceinline__ int fragA(int m, int kk) {
  int hi   = (kk >> 4) & 1;                       // lanes 16-31 hold K 16-31
  int lane = (m & 15) | (hi << 4);
  return lane * 16 + (kk & 15);
}

template<int DIN, int KX, bool WRITE_ALL>
__global__ __launch_bounds__(256, 1)
void lstm_dir_kernel(const float* __restrict__ in,   // [B][S][DIN]
                     const float* __restrict__ wih_f, const float* __restrict__ whh_f,
                     const float* __restrict__ bih_f, const float* __restrict__ bhh_f,
                     const float* __restrict__ wih_r, const float* __restrict__ whh_r,
                     const float* __restrict__ bih_r, const float* __restrict__ bhh_r,
                     float* __restrict__ out)        // WRITE_ALL ? [B][S][128] : [B][128]
{
  const int dir   = blockIdx.y;
  const int t0    = threadIdx.x;          // 0..255 (8 wave32)
  const int bbase = blockIdx.x * BT;

  const float* Wih = dir ? wih_r : wih_f;
  const float* Whh = dir ? whh_r : whh_f;
  const float* Bih = dir ? bih_r : bih_f;
  const float* Bhh = dir ? bhh_r : bhh_f;

  extern __shared__ char smem_raw[];
  _Float16* Xfrag = (_Float16*)smem_raw;               // MT*KX A-fragments (1KB each)
  _Float16* Hfrag = Xfrag + MT * KX * 512;             // MT*KH A-fragments
  float*    G     = (float*)(Hfrag + MT * KH * 512);   // [BT][NGATE] gate pre-activations

  const int wv    = t0 >> 5;
  const int lane  = t0 & 31;
  const int mtile = wv & (MT - 1);
  const int nh    = wv / MT;              // 0..3 -> n-tiles nh*4 .. nh*4+3

  // ---- one-time: weight B-fragments straight from global into registers ----
  // B-frag layout (contiguous k, same as fragA): lane holds col n = ntile*16+(lane&15),
  // halves r=0..15 hold k = ks*32 + (lane&16) + r. Weights are loop-invariant
  // over all 256 timesteps -> keep them register-resident.
  v16h bih[KX][NTPW];
  v16h bhh[KH][NTPW];
  {
    const int n = (lane & 15);
    #pragma unroll
    for (int ks = 0; ks < KX; ++ks)
      #pragma unroll
      for (int nt = 0; nt < NTPW; ++nt) {
        int ncol = ((nh * NTPW + nt) * 16 + n);
        v16h b;
        #pragma unroll
        for (int r = 0; r < 16; ++r) {
          int k = ks * 32 + (lane & 16) + r;
          b[r] = (k < DIN) ? (_Float16)Wih[ncol * DIN + k] : (_Float16)0.0f;
        }
        bih[ks][nt] = b;
      }
    #pragma unroll
    for (int ks = 0; ks < KH; ++ks)
      #pragma unroll
      for (int nt = 0; nt < NTPW; ++nt) {
        int ncol = ((nh * NTPW + nt) * 16 + n);
        v16h b;
        #pragma unroll
        for (int r = 0; r < 16; ++r) {
          int k = ks * 32 + (lane & 16) + r;
          b[r] = (_Float16)Whh[ncol * HID + k];
        }
        bhh[ks][nt] = b;
      }
  }

  for (int idx = t0; idx < MT * KX * 512; idx += 256) Xfrag[idx] = (_Float16)0.0f; // zero-pad K
  for (int idx = t0; idx < MT * KH * 512; idx += 256) Hfrag[idx] = (_Float16)0.0f; // h0 = 0

  // per-thread LSTM-cell constants: this thread owns hidden unit j for rows m = r*4+rb
  const int j  = t0 & 63;
  const int rb = t0 >> 6;
  float bsum[4];
  #pragma unroll
  for (int g = 0; g < 4; ++g) bsum[g] = Bih[g * HID + j] + Bhh[g * HID + j];

  float creg[BT * HID / 256];             // c state lives in registers
  #pragma unroll
  for (int r = 0; r < BT * HID / 256; ++r) creg[r] = 0.0f;

  __syncthreads();

  for (int sstep = 0; sstep < S_LEN; ++sstep) {
    const int t = dir ? (S_LEN - 1 - sstep) : sstep;

    // stage x_t into A-fragment layout (f16)
    for (int e = t0; e < BT * DIN; e += 256) {
      int m = e / DIN, k = e % DIN;
      float xv = in[((size_t)(bbase + m) * S_LEN + t) * DIN + k];
      Xfrag[((m >> 4) * KX + (k >> 5)) * 512 + fragA(m, k & 31)] = (_Float16)xv;
    }
    // pull next timestep's input toward L2 (global_prefetch_b8)
    if (sstep + 1 < S_LEN) {
      int tn = dir ? (S_LEN - 2 - sstep) : (sstep + 1);
      __builtin_prefetch(&in[((size_t)(bbase + (t0 & (BT - 1))) * S_LEN + tn) * DIN], 0, 1);
    }
    __syncthreads();

    // G = x_t @ Wih^T + h @ Whh^T  via v_wmma_f32_16x16x32_f16
    // (B operands already in VGPRs; only A fragments come from LDS)
    {
      v8f acc[NTPW] = {};
      #pragma unroll
      for (int ks = 0; ks < KX; ++ks) {
        v16h a = *(const v16h*)(Xfrag + (mtile * KX + ks) * 512 + lane * 16);
        #pragma unroll
        for (int nt = 0; nt < NTPW; ++nt)
          acc[nt] = __builtin_amdgcn_wmma_f32_16x16x32_f16(false, a, false, bih[ks][nt],
                                                           (short)0, acc[nt], false, false);
      }
      #pragma unroll
      for (int ks = 0; ks < KH; ++ks) {
        v16h a = *(const v16h*)(Hfrag + (mtile * KH + ks) * 512 + lane * 16);
        #pragma unroll
        for (int nt = 0; nt < NTPW; ++nt)
          acc[nt] = __builtin_amdgcn_wmma_f32_16x16x32_f16(false, a, false, bhh[ks][nt],
                                                           (short)0, acc[nt], false, false);
      }
      // C/D layout: VGPR v -> row v (lanes 0-15) / v+8 (lanes 16-31), col = lane&15
      int mrow = mtile * 16 + ((lane & 16) ? 8 : 0);
      #pragma unroll
      for (int nt = 0; nt < NTPW; ++nt) {
        int ncol = (nh * NTPW + nt) * 16 + (lane & 15);
        #pragma unroll
        for (int v = 0; v < 8; ++v)
          G[(mrow + v) * NGATE + ncol] = acc[nt][v];
      }
    }
    __syncthreads();

    // elementwise LSTM cell (fp32); h scattered straight into A-fragment slots
    #pragma unroll
    for (int r = 0; r < BT * HID / 256; ++r) {
      int m = r * 4 + rb;
      float gi = G[m * NGATE + j]             + bsum[0];
      float gf = G[m * NGATE + HID + j]       + bsum[1];
      float gg = G[m * NGATE + 2 * HID + j]   + bsum[2];
      float go = G[m * NGATE + 3 * HID + j]   + bsum[3];
      float c  = sigm(gf) * creg[r] + sigm(gi) * tanh_fast(gg);
      creg[r]  = c;
      float h  = sigm(go) * tanh_fast(c);
      Hfrag[((m >> 4) * KH + (j >> 5)) * 512 + fragA(m, j & 31)] = (_Float16)h;
      if (WRITE_ALL) {
        out[((size_t)(bbase + m) * S_LEN + t) * 128 + dir * HID + j] = h;
      } else if (t == S_LEN - 1) {
        out[(size_t)(bbase + m) * 128 + dir * HID + j] = h;
      }
    }
    __syncthreads();
  }
}

// final FC: out[b, cls] = h1_last[b, :] . fc_w[cls, :] + fc_b[cls]
__global__ void fc_kernel(const float* __restrict__ h1last,  // [1024][128]
                          const float* __restrict__ fc_w,    // [3][128]
                          const float* __restrict__ fc_b,    // [3]
                          float* __restrict__ out)           // [1024][3]
{
  int i = blockIdx.x * blockDim.x + threadIdx.x;
  if (i >= 1024 * 3) return;
  int b = i / 3, cls = i % 3;
  float acc = fc_b[cls];
  #pragma unroll 4
  for (int k = 0; k < 128; ++k) acc += h1last[b * 128 + k] * fc_w[cls * 128 + k];
  out[i] = acc;
}

extern "C" void kernel_launch(void* const* d_in, const int* in_sizes, int n_in,
                              void* d_out, int out_size, void* d_ws, size_t ws_size,
                              hipStream_t stream) {
  (void)in_sizes; (void)n_in; (void)out_size; (void)ws_size;
  const float* x = (const float*)d_in[0];
  float* h0     = (float*)d_ws;                             // [1024][256][128] = 128 MB
  float* h1last = h0 + (size_t)1024 * 256 * 128;            // [1024][128]
  float* out    = (float*)d_out;

  const size_t smem0 = (size_t)((MT * 1 + MT * KH) * 512) * 2 + (size_t)BT * NGATE * 4; // ~38 KB
  const size_t smem1 = (size_t)((MT * 4 + MT * KH) * 512) * 2 + (size_t)BT * NGATE * 4; // ~44 KB

  // layer 0 (DIN=16 zero-padded to one K-step), both directions concurrently
  lstm_dir_kernel<16, 1, true><<<dim3(1024 / BT, 2), 256, smem0, stream>>>(
      x,
      (const float*)d_in[1], (const float*)d_in[2], (const float*)d_in[3], (const float*)d_in[4],
      (const float*)d_in[5], (const float*)d_in[6], (const float*)d_in[7], (const float*)d_in[8],
      h0);
  // layer 1 (DIN=128, 4 K-steps for the input projection); only t = S-1 is kept
  lstm_dir_kernel<128, 4, false><<<dim3(1024 / BT, 2), 256, smem1, stream>>>(
      h0,
      (const float*)d_in[9],  (const float*)d_in[10], (const float*)d_in[11], (const float*)d_in[12],
      (const float*)d_in[13], (const float*)d_in[14], (const float*)d_in[15], (const float*)d_in[16],
      h1last);
  fc_kernel<<<(1024 * 3 + 255) / 256, 256, 0, stream>>>(
      h1last, (const float*)d_in[17], (const float*)d_in[18], out);
}
